// MultiHeadAttention_67972152426603
// MI455X (gfx1250) — matrix operand order repaired and checked by reading, not compiled, not run on hardware
//
#include <hip/hip_runtime.h>
#include <hip/hip_bf16.h>
#include <cstdint>

// ---------------------------------------------------------------------------
// MHA forward for MI455X (gfx1250, wave32, WMMA f32_16x16x32_f16)
// B=2, L=2048, D=1024, H=16, DK=64
// ---------------------------------------------------------------------------

#define B_   2
#define L_   2048
#define D_   1024
#define H_   16
#define DK_  64

typedef __attribute__((ext_vector_type(16))) _Float16 v16h;
typedef __attribute__((ext_vector_type(8)))  _Float16 v8h;
typedef __attribute__((ext_vector_type(8)))  float    v8f;

union V16HU { v16h v; v8h h[2]; };

__device__ inline v8f wmma_f16(v16h a, v16h b, v8f c) {
  // v_wmma_f32_16x16x32_f16  D = A(16x32) * B(32x16) + C(16x16 f32)
  return __builtin_amdgcn_wmma_f32_16x16x32_f16(
      /*neg_a=*/false, a, /*neg_b=*/false, b,
      /*c_mod=*/(short)0, c, /*reuse_a=*/false, /*reuse_b=*/false);
}

// A-fragment: 16x32 (MxK) f16.  Lane l: row m=l&15; halves e0..7 = K[8*kh .. 8*kh+7],
// e8..15 = K[16+8*kh .. 23+8*kh] with kh = l>>4.  Source row-major, K contiguous.
__device__ inline v16h load_a_frag(const _Float16* base, int stride, int kbase) {
  const int lane = threadIdx.x & 31;
  const int m    = lane & 15;
  const int kh8  = (lane >> 4) * 8;
  const _Float16* p = base + (size_t)m * stride + kbase + kh8;
  V16HU u;
  u.h[0] = *(const v8h*)(p);
  u.h[1] = *(const v8h*)(p + 16);
  return u.v;
}

// B-fragment: 32x16 (KxN) f16 where B[k][n] = S[n][k], S row-major (K contiguous).
// Lane l: col n=l&15; e0..7 = K[8*kh .. 8*kh+7], e8..15 = K[16+8*kh .. 23+8*kh], kh = l>>4.
__device__ inline v16h load_b_frag(const _Float16* base, int stride, int kbase) {
  const int lane = threadIdx.x & 31;
  const int n    = lane & 15;
  const int kh8  = (lane >> 4) * 8;
  const _Float16* p = base + (size_t)n * stride + kbase + kh8;
  V16HU u;
  u.h[0] = *(const v8h*)(p);
  u.h[1] = *(const v8h*)(p + 16);
  return u.v;
}

// ---------------------------------------------------------------------------
__global__ void cvt_f32_to_f16(const float* __restrict__ src,
                               _Float16* __restrict__ dst, int n) {
  int i = blockIdx.x * blockDim.x + threadIdx.x;
  if (i < n) dst[i] = (_Float16)src[i];
}

// ---------------------------------------------------------------------------
// C[M,1024] = X[M,1024] @ W[1024,1024]^T + bias.  One wave -> 16x64 output tile.
// Software-pipelined: next k-step fragments loaded before current WMMAs issue.
// mode 0: clamp, f16 head-split [B,H,L,DK]   (q, k)
// mode 1: clamp, f16 transposed [B,H,DK,L]   (v)
// mode 2: f32 plain [M,1024] to d_out        (final projection)
__global__ __launch_bounds__(128)
void gemm_wmma(const _Float16* __restrict__ X,
               const _Float16* __restrict__ W,
               const float*    __restrict__ bias,
               _Float16* __restrict__ out_h,
               float*    __restrict__ out_f,
               int M, int mode) {
  const int wid = blockIdx.x * (blockDim.x >> 5) + (threadIdx.x >> 5);
  const int ntiles = D_ / 64;              // 16
  const int tm = wid / ntiles;
  const int tn = wid % ntiles;
  if (tm * 16 >= M) return;                // wave-uniform

  const int lane  = threadIdx.x & 31;
  const int nloc  = lane & 15;
  const int rbase = (lane >> 4) * 8;

  v8f acc[4] = {v8f{}, v8f{}, v8f{}, v8f{}};
  const _Float16* xrow = X + (size_t)(tm * 16) * D_;
  const _Float16* wbase = W + (size_t)(tn * 64) * D_;

  // prologue: fragments for k = 0
  v16h a_cur = load_a_frag(xrow, D_, 0);
  v16h b_cur0 = load_b_frag(wbase + (size_t)0  * D_, D_, 0);
  v16h b_cur1 = load_b_frag(wbase + (size_t)16 * D_, D_, 0);
  v16h b_cur2 = load_b_frag(wbase + (size_t)32 * D_, D_, 0);
  v16h b_cur3 = load_b_frag(wbase + (size_t)48 * D_, D_, 0);

  for (int k = 0; k < D_; k += 32) {
    const int kn = (k + 32) & (D_ - 1);    // wraps on last iter: redundant, branch-free
    // issue next-step loads BEFORE consuming current fragments
    v16h a_nxt  = load_a_frag(xrow, D_, kn);
    v16h b_nxt0 = load_b_frag(wbase + (size_t)0  * D_, D_, kn);
    v16h b_nxt1 = load_b_frag(wbase + (size_t)16 * D_, D_, kn);
    v16h b_nxt2 = load_b_frag(wbase + (size_t)32 * D_, D_, kn);
    v16h b_nxt3 = load_b_frag(wbase + (size_t)48 * D_, D_, kn);

    acc[0] = wmma_f16(a_cur, b_cur0, acc[0]);
    acc[1] = wmma_f16(a_cur, b_cur1, acc[1]);
    acc[2] = wmma_f16(a_cur, b_cur2, acc[2]);
    acc[3] = wmma_f16(a_cur, b_cur3, acc[3]);

    a_cur = a_nxt;
    b_cur0 = b_nxt0; b_cur1 = b_nxt1; b_cur2 = b_nxt2; b_cur3 = b_nxt3;
  }

#pragma unroll
  for (int t = 0; t < 4; ++t) {
    const int n_glob = tn * 64 + t * 16 + nloc;
    const float bn = bias[n_glob];
#pragma unroll
    for (int r = 0; r < 8; ++r) {
      const int m_glob = tm * 16 + rbase + r;
      float v = acc[t][r] + bn;
      if (mode == 2) {
        out_f[(size_t)m_glob * D_ + n_glob] = v;
      } else {
        v = fminf(fmaxf(v, -1.0e4f), 1.0e4f);      // nan_to_num clamp
        const int b  = m_glob / L_;
        const int lr = m_glob % L_;
        const int h  = n_glob / DK_;
        const int dk = n_glob % DK_;
        if (mode == 0)
          out_h[((size_t)(b * H_ + h) * L_ + lr) * DK_ + dk] = (_Float16)v;
        else
          out_h[((size_t)(b * H_ + h) * DK_ + dk) * L_ + lr] = (_Float16)v;
      }
    }
  }
}

// ---------------------------------------------------------------------------
// Flash attention: one wave handles a 16-row q tile for one (b,h).
// q,k: [B,H,L,DK] f16 ; vt: [B,H,DK,L] f16 ; bias: [B,H,L,L] f32
// out: [B*L, D] f16 (heads recombined)
// All global loads of an iteration (K frags, V frags, bias) are issued up
// front so they stream under the QK^T WMMAs and the softmax VALU work.
__global__ __launch_bounds__(128)
void attn_wmma(const _Float16* __restrict__ qhp,
               const _Float16* __restrict__ khp,
               const _Float16* __restrict__ vtp,
               const float*    __restrict__ bias,
               _Float16* __restrict__ attn_out) {
  __shared__ __align__(16) _Float16 pstage[4][16 * 40];  // 16x32 P tile, padded rows

  const int wslot = threadIdx.x >> 5;
  const int wid   = blockIdx.x * 4 + wslot;
  const int qtiles = L_ / 16;                 // 128
  const int qt = wid % qtiles;
  const int bh = wid / qtiles;                // 0 .. B*H-1
  const int b  = bh / H_;
  const int h  = bh % H_;

  const int lane  = threadIdx.x & 31;
  const int nloc  = lane & 15;
  const int rbase = (lane >> 4) * 8;

  const _Float16* qp = qhp + (size_t)bh * L_ * DK_;
  const _Float16* kp = khp + (size_t)bh * L_ * DK_;
  const _Float16* vp = vtp + (size_t)bh * DK_ * L_;
  const float*    bp = bias + (size_t)bh * L_ * L_ + (size_t)(qt * 16) * L_;

  // Q A-fragments (held in registers for the whole key loop)
  const v16h aq0 = load_a_frag(qp + (size_t)(qt * 16) * DK_, DK_, 0);
  const v16h aq1 = load_a_frag(qp + (size_t)(qt * 16) * DK_, DK_, 32);

  v8f o0 = {}, o1 = {}, o2 = {}, o3 = {};
  float mrow[8], lrow[8];
#pragma unroll
  for (int r = 0; r < 8; ++r) { mrow[r] = -3.0e38f; lrow[r] = 0.0f; }

  const float scale = 0.125f;                 // 1/sqrt(DK)
  _Float16* pl = pstage[wslot];

  for (int cb = 0; cb < L_; cb += 32) {
    // ---- issue ALL global loads for this iteration first ----
    v16h b00 = load_b_frag(kp + (size_t)cb * DK_,        DK_, 0);
    v16h b01 = load_b_frag(kp + (size_t)cb * DK_,        DK_, 32);
    v16h b10 = load_b_frag(kp + (size_t)(cb + 16) * DK_, DK_, 0);
    v16h b11 = load_b_frag(kp + (size_t)(cb + 16) * DK_, DK_, 32);

    v16h bv0 = load_b_frag(vp + (size_t)0  * L_ + cb, L_, 0);   // independent of P
    v16h bv1 = load_b_frag(vp + (size_t)16 * L_ + cb, L_, 0);
    v16h bv2 = load_b_frag(vp + (size_t)32 * L_ + cb, L_, 0);
    v16h bv3 = load_b_frag(vp + (size_t)48 * L_ + cb, L_, 0);

    float bz0[8], bz1[8];
#pragma unroll
    for (int r = 0; r < 8; ++r) {
      bz0[r] = bp[(size_t)(rbase + r) * L_ + cb + nloc];
      bz1[r] = bp[(size_t)(rbase + r) * L_ + cb + 16 + nloc];
    }
    if (cb + 32 < L_)   // prefetch next bias tile (-> global_prefetch)
      __builtin_prefetch((const void*)(bp + (size_t)rbase * L_ + cb + 32), 0, 1);

    // ---- logits tile S = (Q K^T) * scale + clipped bias : 16 x 32 ----
    v8f s0 = {}, s1 = {};
    s0 = wmma_f16(aq0, b00, s0);
    s0 = wmma_f16(aq1, b01, s0);
    s1 = wmma_f16(aq0, b10, s1);
    s1 = wmma_f16(aq1, b11, s1);

    // ---- online softmax update (row stats live in component r) ----
#pragma unroll
    for (int r = 0; r < 8; ++r) {
      float bv0c = fminf(fmaxf(bz0[r], -1.0e4f), 1.0e4f);
      float bv1c = fminf(fmaxf(bz1[r], -1.0e4f), 1.0e4f);
      float x0 = s0[r] * scale + bv0c;
      float x1 = s1[r] * scale + bv1c;

      float bm = fmaxf(x0, x1);               // reduce over 16 cols (half-group)
      bm = fmaxf(bm, __shfl_xor(bm, 1, 32));
      bm = fmaxf(bm, __shfl_xor(bm, 2, 32));
      bm = fmaxf(bm, __shfl_xor(bm, 4, 32));
      bm = fmaxf(bm, __shfl_xor(bm, 8, 32));

      float mnew  = fmaxf(mrow[r], bm);
      float alpha = __expf(mrow[r] - mnew);
      float e0 = __expf(x0 - mnew);
      float e1 = __expf(x1 - mnew);

      float ps = e0 + e1;
      ps += __shfl_xor(ps, 1, 32);
      ps += __shfl_xor(ps, 2, 32);
      ps += __shfl_xor(ps, 4, 32);
      ps += __shfl_xor(ps, 8, 32);

      lrow[r] = lrow[r] * alpha + ps;
      mrow[r] = mnew;
      o0[r] *= alpha; o1[r] *= alpha; o2[r] *= alpha; o3[r] *= alpha;

      // stage P (f16, row-major 16x32, padded stride 40) for re-fragmenting
      pl[(rbase + r) * 40 + nloc]      = (_Float16)e0;
      pl[(rbase + r) * 40 + 16 + nloc] = (_Float16)e1;
    }

    // ---- O += P @ V  (A = P from LDS, B = V^T rows already in regs) ----
    v16h ap;
    {
      const int kh8 = (lane >> 4) * 8;
      V16HU u;
      u.h[0] = *(const v8h*)(&pl[nloc * 40 + kh8]);
      u.h[1] = *(const v8h*)(&pl[nloc * 40 + kh8 + 16]);
      ap = u.v;
    }
    o0 = wmma_f16(ap, bv0, o0);
    o1 = wmma_f16(ap, bv1, o1);
    o2 = wmma_f16(ap, bv2, o2);
    o3 = wmma_f16(ap, bv3, o3);
  }

  // ---- normalize and write recombined [B*L, D] f16 ----
#pragma unroll
  for (int r = 0; r < 8; ++r) {
    const float inv = lrow[r] > 0.0f ? 1.0f / lrow[r] : 0.0f;
    const int m_glob = b * L_ + qt * 16 + rbase + r;
    _Float16* orow = attn_out + (size_t)m_glob * D_ + h * DK_;
    orow[0 * 16 + nloc] = (_Float16)(o0[r] * inv);
    orow[1 * 16 + nloc] = (_Float16)(o1[r] * inv);
    orow[2 * 16 + nloc] = (_Float16)(o2[r] * inv);
    orow[3 * 16 + nloc] = (_Float16)(o3[r] * inv);
  }
}

// ---------------------------------------------------------------------------
extern "C" void kernel_launch(void* const* d_in, const int* in_sizes, int n_in,
                              void* d_out, int out_size, void* d_ws, size_t ws_size,
                              hipStream_t stream) {
  const float* Q    = (const float*)d_in[0];
  const float* K    = (const float*)d_in[1];
  const float* V    = (const float*)d_in[2];
  const float* bias = (const float*)d_in[3];
  const float* Wq   = (const float*)d_in[4];
  const float* bq   = (const float*)d_in[5];
  const float* Wk   = (const float*)d_in[6];
  const float* bk   = (const float*)d_in[7];
  const float* Wv   = (const float*)d_in[8];
  const float* bv   = (const float*)d_in[9];
  const float* Wo   = (const float*)d_in[10];
  const float* bo   = (const float*)d_in[11];
  float* out = (float*)d_out;

  const size_t WSZ = (size_t)D_ * D_;        // 1,048,576 halves
  const size_t XSZ = (size_t)B_ * L_ * D_;   // 4,194,304 halves
  _Float16* w    = (_Float16*)d_ws;
  _Float16* wq_h = w;
  _Float16* wk_h = wq_h + WSZ;
  _Float16* wv_h = wk_h + WSZ;
  _Float16* wo_h = wv_h + WSZ;
  _Float16* q_in = wo_h + WSZ;
  _Float16* k_in = q_in + XSZ;
  _Float16* v_in = k_in + XSZ;
  _Float16* qh   = v_in + XSZ;
  _Float16* kh   = qh + XSZ;
  _Float16* vt   = kh + XSZ;
  _Float16* attn = vt + XSZ;                 // total ~64 MB

  const int CT = 256;
  cvt_f32_to_f16<<<(int)(WSZ / CT), CT, 0, stream>>>(Wq, wq_h, (int)WSZ);
  cvt_f32_to_f16<<<(int)(WSZ / CT), CT, 0, stream>>>(Wk, wk_h, (int)WSZ);
  cvt_f32_to_f16<<<(int)(WSZ / CT), CT, 0, stream>>>(Wv, wv_h, (int)WSZ);
  cvt_f32_to_f16<<<(int)(WSZ / CT), CT, 0, stream>>>(Wo, wo_h, (int)WSZ);
  cvt_f32_to_f16<<<(int)(XSZ / CT), CT, 0, stream>>>(Q, q_in, (int)XSZ);
  cvt_f32_to_f16<<<(int)(XSZ / CT), CT, 0, stream>>>(K, k_in, (int)XSZ);
  cvt_f32_to_f16<<<(int)(XSZ / CT), CT, 0, stream>>>(V, v_in, (int)XSZ);

  const int M = B_ * L_;                                  // 4096
  const int gemm_blocks = (M / 16) * (D_ / 64) / 4;       // 1024 (4 waves/block)
  gemm_wmma<<<gemm_blocks, 128, 0, stream>>>(q_in, wq_h, bq, qh, nullptr, M, 0);
  gemm_wmma<<<gemm_blocks, 128, 0, stream>>>(k_in, wk_h, bk, kh, nullptr, M, 0);
  gemm_wmma<<<gemm_blocks, 128, 0, stream>>>(v_in, wv_h, bv, vt, nullptr, M, 1);

  const int attn_blocks = B_ * H_ * (L_ / 16) / 4;        // 1024
  attn_wmma<<<attn_blocks, 128, 0, stream>>>(qh, kh, vt, bias, attn);

  gemm_wmma<<<gemm_blocks, 128, 0, stream>>>(attn, wo_h, bo, nullptr, out, M, 2);
}
